// SelfAttention_644245094748
// MI455X (gfx1250) — matrix operand order repaired and checked by reading, not compiled, not run on hardware
//
#include <hip/hip_runtime.h>
#include <stdint.h>

typedef __attribute__((ext_vector_type(16))) _Float16 v16h;
typedef __attribute__((ext_vector_type(8)))  _Float16 v8h;
typedef __attribute__((ext_vector_type(8)))  float    v8f;

// Problem constants (from reference)
constexpr int BB = 2, C = 512, S = 2304;             // H*W = 48*48
constexpr int NH = 8, HD = 64, G = 32, CPG = C / G;  // 8 heads of 64, 32 groups of 16
constexpr float EPSV  = 1e-5f;
constexpr float SCALE = 0.04419417382415922f;        // 1/sqrt(512)

#define DEVI __device__ __forceinline__

DEVI v16h cat8(v8h lo, v8h hi) {
  v16h r;
#pragma unroll
  for (int i = 0; i < 8; ++i) { r[i] = lo[i]; r[i + 8] = hi[i]; }
  return r;
}

// A fragment (16x32 f16, MxK): lanes 0-15 hold row M=lane, K 0..15;
// lanes 16-31 hold row M=lane-16, K 16..31. src row-major [row][k].
DEVI v16h load_a_frag(const _Float16* src, int row, int ld, int koff) {
  int lane = threadIdx.x & 31;
  const _Float16* p = src + (row + (lane & 15)) * ld + koff + ((lane >> 4) << 4);
  v8h lo = *(const v8h*)p;
  v8h hi = *(const v8h*)(p + 8);
  return cat8(lo, hi);
}

// B fragment (32x16 f16, KxN): lane holds col N=lane&15, contiguous K run
// (lanes 0-15: K 0..15, lanes 16-31: K 16..31). srcT is [n][k] row-major.
DEVI v16h load_b_frag(const _Float16* srcT, int col, int ld, int koff) {
  int lane = threadIdx.x & 31;
  const _Float16* p = srcT + (col + (lane & 15)) * ld + koff + ((lane >> 4) << 4);
  v8h lo = *(const v8h*)p;
  v8h hi = *(const v8h*)(p + 8);
  return cat8(lo, hi);
}

DEVI v8f wmma16(v16h a, v16h b, v8f c) {
  return __builtin_amdgcn_wmma_f32_16x16x32_f16(false, a, false, b, (short)0, c,
                                                false, false);
}

// CDNA5 async copy: global -> LDS, 16B per lane, tracked by ASYNCcnt.
DEVI void async_b128(const void* g, const void* l) {
  unsigned lds_off = (unsigned)(uintptr_t)l;           // low 32b = LDS offset
  unsigned long long ga = (unsigned long long)(uintptr_t)g;
  asm volatile("global_load_async_to_lds_b128 %0, %1, off"
               :: "v"(lds_off), "v"(ga) : "memory");
}
DEVI void wait_async() { asm volatile("s_wait_asynccnt 0x0" ::: "memory"); }

// ---------------- GroupNorm ----------------
__global__ __launch_bounds__(256) void gn_stats_k(const float* __restrict__ x,
                                                  float* __restrict__ stats) {
  __shared__ float s_sum[256], s_ss[256];
  const int tid = threadIdx.x;
  const int bg = blockIdx.x;            // b*32 + g
  const int b = bg >> 5, g = bg & 31;
  const float* p = x + ((size_t)b * C + g * CPG) * S;
  const int cnt = CPG * S;              // 36864
  float sum = 0.f, ss = 0.f;
  for (int i = tid; i < cnt; i += 256) { float v = p[i]; sum += v; ss += v * v; }
  s_sum[tid] = sum; s_ss[tid] = ss;
  __syncthreads();
  for (int off = 128; off > 0; off >>= 1) {
    if (tid < off) { s_sum[tid] += s_sum[tid + off]; s_ss[tid] += s_ss[tid + off]; }
    __syncthreads();
  }
  if (tid == 0) {
    float mean = s_sum[0] / (float)cnt;
    float var = s_ss[0] / (float)cnt - mean * mean;
    stats[2 * bg] = mean;
    stats[2 * bg + 1] = rsqrtf(var + EPSV);
  }
}

// Writes xn TRANSPOSED: xnT[b][s][c] (f16) so GEMM-1 B-staging is a plain copy.
__global__ __launch_bounds__(256) void gn_norm_k(const float* __restrict__ x,
                                                 const float* __restrict__ gamma,
                                                 const float* __restrict__ beta,
                                                 const float* __restrict__ stats,
                                                 _Float16* __restrict__ xnT) {
  int i = blockIdx.x * 256 + threadIdx.x;          // B*S*C, c fastest (coalesced store)
  int c = i & (C - 1);
  int s = (i >> 9) % S;
  int b = i / (C * S);
  int bg = b * 32 + (c >> 4);
  float mean = stats[2 * bg], rstd = stats[2 * bg + 1];
  float v = (x[((size_t)(b * C + c)) * S + s] - mean) * rstd * gamma[c] + beta[c];
  xnT[i] = (_Float16)v;
}

// ---------------- weight casts (fold score scale into Q rows) ----------------
__global__ __launch_bounds__(256) void cvt_wqkv_k(const float* __restrict__ w,
                                                  _Float16* __restrict__ o) {
  int i = blockIdx.x * 256 + threadIdx.x;          // 1536*512
  int oc = i >> 9;
  float s = ((oc % (3 * HD)) < HD) ? SCALE : 1.0f;
  o[i] = (_Float16)(w[i] * s);
}

__global__ __launch_bounds__(256) void cvt_wout_k(const float* __restrict__ w,
                                                  _Float16* __restrict__ o) {
  int i = blockIdx.x * 256 + threadIdx.x;          // 512*512
  o[i] = (_Float16)w[i];
}

// ---------------- WMMA GEMM: out[b][m][n] = sum_k W[m][k] * XT[b][n][k] ----------------
// 256 thr = 8 waves (2x4), tile 128x128, K staged 32/step, async double-buffered LDS.
// !FUSE (QKV): q/k rows -> qkT[b][s][h*128 + qk*64 + d] (transposed, vector stores),
//              v rows   -> vbuf[b][h*64+d][s].
// FUSE (proj): fp32 out[b][c][s] = acc + bias[c] + resid[b][c][s].
template <bool FUSE>
__global__ __launch_bounds__(256) void gemm_k(const _Float16* __restrict__ W,
                                              const _Float16* __restrict__ XT,
                                              _Float16* __restrict__ outQK,
                                              _Float16* __restrict__ outV,
                                              float* __restrict__ outF,
                                              const float* __restrict__ bias,
                                              const float* __restrict__ resid) {
  __shared__ _Float16 As[2][128 * 40];   // [m][k], ld 40
  __shared__ _Float16 Bs[2][128 * 40];   // [n][k], ld 40
  const int tid = threadIdx.x;
  const int lane = tid & 31;
  const int wv = tid >> 5;
  const int wm = wv >> 2, wn = wv & 3;
  const int ln = lane & 15, gl = lane >> 4;
  const int nb = blockIdx.x * 128;
  const int mb = blockIdx.y * 128;
  const int bz = blockIdx.z;
  const _Float16* XTb = XT + (size_t)bz * S * C;

  v8f acc[4][2];
#pragma unroll
  for (int mf = 0; mf < 4; ++mf)
#pragma unroll
    for (int nf = 0; nf < 2; ++nf)
#pragma unroll
      for (int i = 0; i < 8; ++i) acc[mf][nf][i] = 0.f;

  const int row = tid >> 2, cg = (tid & 3) * 8;        // 64 rows per pass, 2 passes
  auto stage = [&](int buf, int k0) {
#pragma unroll
    for (int it = 0; it < 2; ++it) {
      int r2 = row + 64 * it;
      async_b128(W + (size_t)(mb + r2) * C + k0 + cg, &As[buf][r2 * 40 + cg]);
      async_b128(XTb + (size_t)(nb + r2) * C + k0 + cg, &Bs[buf][r2 * 40 + cg]);
    }
  };

  stage(0, 0);
  wait_async();
  __syncthreads();

  for (int ks = 0; ks < 16; ++ks) {
    const int cur = ks & 1;
    if (ks + 1 < 16) stage(cur ^ 1, (ks + 1) * 32);    // overlap DMA with WMMA

    v16h bfr[2];
#pragma unroll
    for (int nf = 0; nf < 2; ++nf)
      bfr[nf] = load_b_frag(Bs[cur], 32 * wn + 16 * nf, 40, 0);
#pragma unroll
    for (int mf = 0; mf < 4; ++mf) {
      v16h af = load_a_frag(As[cur], 64 * wm + 16 * mf, 40, 0);
#pragma unroll
      for (int nf = 0; nf < 2; ++nf) acc[mf][nf] = wmma16(af, bfr[nf], acc[mf][nf]);
    }
    wait_async();            // next-buffer DMA landed
    __syncthreads();         // all waves done reading cur before overwrite
  }

#pragma unroll
  for (int mf = 0; mf < 4; ++mf) {
#pragma unroll
    for (int nf = 0; nf < 2; ++nf) {
      int n = nb + 32 * wn + 16 * nf + ln;
      int mbase = mb + 64 * wm + 16 * mf;              // 16-aligned
      if constexpr (FUSE) {
#pragma unroll
        for (int r = 0; r < 8; ++r) {
          int m = mbase + 8 * gl + r;                  // C/D layout: M = r + 8*lanegrp
          size_t idx = (size_t)(bz * C + m) * S + n;
          outF[idx] = acc[mf][nf][r] + bias[m] + resid[idx];
        }
      } else {
        int h = mbase / 192, r192 = mbase % 192;
        if (r192 < 128) {                              // q or k rows: transposed store
          v8h o;
#pragma unroll
          for (int r = 0; r < 8; ++r) o[r] = (_Float16)acc[mf][nf][r];
          *(v8h*)(outQK + ((size_t)(bz * S + n)) * 1024 + h * 128 + r192 + 8 * gl) = o;
        } else {                                        // v rows: row-major store
          int c = h * 64 + (r192 - 128) + 8 * gl;
#pragma unroll
          for (int r = 0; r < 8; ++r)
            outV[((size_t)(bz * C + c + r)) * S + n] = (_Float16)acc[mf][nf][r];
        }
      }
    }
  }
}

// ---------------- Flash attention, register-resident online softmax ----------------
// One block (4 waves) per (b, head, 64-query tile). k-chunks async double-buffered.
__global__ __launch_bounds__(128) void attn_k(const _Float16* __restrict__ qkT,
                                              const _Float16* __restrict__ vbuf,
                                              _Float16* __restrict__ attnT) {
  __shared__ _Float16 qT[64 * 72];      // [s][d]
  __shared__ _Float16 kT[2][64 * 72];   // [t][d], double-buffered
  __shared__ _Float16 p16[64 * 72];     // [s][t], wave-private row strips

  const int tid = threadIdx.x;
  const int lane = tid & 31;
  const int wv = tid >> 5;
  const int ln = lane & 15, gl = lane >> 4;
  const int s0 = blockIdx.x * 64;
  const int h = blockIdx.y;
  const int b = blockIdx.z;
  const _Float16* qkTb = qkT + (size_t)b * S * 1024 + h * 128;
  const _Float16* vb = vbuf + ((size_t)(b * C + h * HD)) * S;

  // stage q^T tile (plain contiguous async copies)
#pragma unroll
  for (int it = 0; it < 4; ++it) {
    int L = tid + 128 * it;                            // 0..511
    int i = L >> 3, ig = (L & 7) * 8;
    async_b128(qkTb + (size_t)(s0 + i) * 1024 + ig, &qT[i * 72 + ig]);
  }
  wait_async();
  __syncthreads();

  v16h aq0 = load_a_frag(qT, 16 * wv, 72, 0);
  v16h aq1 = load_a_frag(qT, 16 * wv, 72, 32);

  float mrun[8], lrun[8];
#pragma unroll
  for (int r = 0; r < 8; ++r) { mrun[r] = -1e30f; lrun[r] = 0.f; }
  v8f acc[4];
#pragma unroll
  for (int nf = 0; nf < 4; ++nf)
#pragma unroll
    for (int i = 0; i < 8; ++i) acc[nf][i] = 0.f;

  auto stage_kT = [&](int buf, int t0) {
#pragma unroll
    for (int it = 0; it < 4; ++it) {
      int L = tid + 128 * it;
      int i = L >> 3, ig = (L & 7) * 8;
      async_b128(qkTb + (size_t)(t0 + i) * 1024 + 64 + ig, &kT[buf][i * 72 + ig]);
    }
  };

  stage_kT(0, 0);
  wait_async();
  __syncthreads();

  const int NCHUNK = S / 64;                           // 36
  for (int ci = 0; ci < NCHUNK; ++ci) {
    const int cur = ci & 1;
    if (ci + 1 < NCHUNK) stage_kT(cur ^ 1, (ci + 1) * 64);  // DMA overlaps compute

    // scores: S[s][t] = sum_d qT[s][d] * kT[t][d]
    v8f sf[4];
#pragma unroll
    for (int nf = 0; nf < 4; ++nf) {
      v16h bk0 = load_b_frag(kT[cur], 16 * nf, 72, 0);
      v16h bk1 = load_b_frag(kT[cur], 16 * nf, 72, 32);
#pragma unroll
      for (int i = 0; i < 8; ++i) sf[nf][i] = 0.f;
      sf[nf] = wmma16(aq0, bk0, sf[nf]);
      sf[nf] = wmma16(aq1, bk1, sf[nf]);
    }

    // register online softmax: rows live per accumulator element, cols across 16 lanes
    float cm[8], fac[8], ps[8];
#pragma unroll
    for (int r = 0; r < 8; ++r) {
      cm[r] = fmaxf(fmaxf(sf[0][r], sf[1][r]), fmaxf(sf[2][r], sf[3][r]));
#pragma unroll
      for (int m = 1; m <= 8; m <<= 1) cm[r] = fmaxf(cm[r], __shfl_xor(cm[r], m, 32));
      float mn = fmaxf(mrun[r], cm[r]);
      fac[r] = __expf(mrun[r] - mn);
      mrun[r] = mn;
      ps[r] = 0.f;
    }
#pragma unroll
    for (int nf = 0; nf < 4; ++nf)
#pragma unroll
      for (int r = 0; r < 8; ++r) {
        float p = __expf(sf[nf][r] - mrun[r]);
        ps[r] += p;
        p16[(16 * wv + 8 * gl + r) * 72 + 16 * nf + ln] = (_Float16)p;
      }
#pragma unroll
    for (int r = 0; r < 8; ++r) {
#pragma unroll
      for (int m = 1; m <= 8; m <<= 1) ps[r] += __shfl_xor(ps[r], m, 32);
      lrun[r] = lrun[r] * fac[r] + ps[r];
    }

    // rescale accumulators + acc += P * V^T  (V B-frags straight from global)
    v16h ap0 = load_a_frag(p16, 16 * wv, 72, 0);   // wave-private strip, DScnt-ordered
    v16h ap1 = load_a_frag(p16, 16 * wv, 72, 32);
    const int t0 = ci * 64;
#pragma unroll
    for (int nf = 0; nf < 4; ++nf) {
#pragma unroll
      for (int r = 0; r < 8; ++r) acc[nf][r] *= fac[r];
      const _Float16* vp = vb + (size_t)(16 * nf + ln) * S + t0 + gl * 16;
      v16h bv0 = cat8(*(const v8h*)vp, *(const v8h*)(vp + 8));
      v16h bv1 = cat8(*(const v8h*)(vp + 32), *(const v8h*)(vp + 40));
      acc[nf] = wmma16(ap0, bv0, acc[nf]);
      acc[nf] = wmma16(ap1, bv1, acc[nf]);
    }

    wait_async();            // next kT buffer landed
    __syncthreads();         // all waves done with kT[cur] before it's overwritten
  }

  // finalize: divide by row sums, emit attnT[b][s][c] for the projection GEMM
  float li[8];
#pragma unroll
  for (int r = 0; r < 8; ++r) li[r] = 1.0f / lrun[r];
#pragma unroll
  for (int nf = 0; nf < 4; ++nf) {
    int d = 16 * nf + ln;
#pragma unroll
    for (int r = 0; r < 8; ++r) {
      int s = s0 + 16 * wv + 8 * gl + r;
      attnT[((size_t)(b * S + s)) * C + h * HD + d] = (_Float16)(acc[nf][r] * li[r]);
    }
  }
}

// ---------------- host ----------------
extern "C" void kernel_launch(void* const* d_in, const int* in_sizes, int n_in,
                              void* d_out, int out_size, void* d_ws, size_t ws_size,
                              hipStream_t stream) {
  const float* x     = (const float*)d_in[0];
  const float* gamma = (const float*)d_in[1];
  const float* beta  = (const float*)d_in[2];
  const float* w_qkv = (const float*)d_in[3];
  const float* w_out = (const float*)d_in[4];
  const float* b_out = (const float*)d_in[5];
  float* out = (float*)d_out;

  char* ws = (char*)d_ws;
  _Float16* xnT   = (_Float16*)(ws);                 // B*S*C f16      = 4,718,592 B
  _Float16* wqh   = (_Float16*)(ws + 4718592);       // 1536*512 f16   = 1,572,864 B
  _Float16* woh   = (_Float16*)(ws + 6291456);       // 512*512 f16    =   524,288 B
  _Float16* qkT   = (_Float16*)(ws + 6815744);       // B*S*1024 f16   = 9,437,184 B
  _Float16* vbuf  = (_Float16*)(ws + 16252928);      // B*512*S f16    = 4,718,592 B
  _Float16* attnT = (_Float16*)(ws + 20971520);      // B*S*512 f16    = 4,718,592 B
  float*    stats = (float*)(ws + 25690112);         // 64*2 f32

  gn_stats_k<<<BB * G, 256, 0, stream>>>(x, stats);
  gn_norm_k<<<(BB * C * S) / 256, 256, 0, stream>>>(x, gamma, beta, stats, xnT);
  cvt_wqkv_k<<<(3 * C * C) / 256, 256, 0, stream>>>(w_qkv, wqh);
  cvt_wout_k<<<(C * C) / 256, 256, 0, stream>>>(w_out, woh);
  gemm_k<false><<<dim3(S / 128, 1536 / 128, BB), 256, 0, stream>>>(
      wqh, xnT, qkT, vbuf, nullptr, nullptr, nullptr);
  attn_k<<<dim3(S / 64, NH, BB), 128, 0, stream>>>(qkT, vbuf, attnT);
  gemm_k<true><<<dim3(S / 128, C / 128, BB), 256, 0, stream>>>(
      woh, attnT, nullptr, nullptr, out, b_out, x);
}
